// MultiHeadAttention_30485677867214
// MI455X (gfx1250) — matrix operand order repaired and checked by reading, not compiled, not run on hardware
//
#include <hip/hip_runtime.h>

// ---------------------------------------------------------------------------
// MHA forward for MI455X (gfx1250, wave32, WMMA + async LDS staging).
//   - all GEMMs: v_wmma_f32_16x16x32_f16, f32 accumulate
//   - weight panels staged to LDS with global_load_async_to_lds_b128,
//     double-buffered, synchronized with s_wait_asynccnt + barriers
// ---------------------------------------------------------------------------

typedef __attribute__((ext_vector_type(16))) _Float16 v16h;
typedef __attribute__((ext_vector_type(8)))  _Float16 v8h;
typedef __attribute__((ext_vector_type(8)))  float    v8f;

#define NEG_INF_F (-3.4028234663852886e38f)

static constexpr int B_  = 4;
static constexpr int L_  = 1024;
static constexpr int D_  = 1024;
static constexpr int H_  = 16;
static constexpr int DK_ = 64;

// ---- CDNA5 async copy primitives ------------------------------------------
static __device__ inline void async_load_b128(unsigned lds_off_bytes,
                                              const void* gaddr) {
  // GLOBAL_LOAD_ASYNC_TO_LDS_B128: copies 16 B from global to LDS at vdst; tracked by ASYNCcnt
  asm volatile("global_load_async_to_lds_b128 %0, %1, off"
               :: "v"(lds_off_bytes), "v"(gaddr)
               : "memory");
}
#define S_WAIT_ASYNC_LE2() asm volatile("s_wait_asynccnt 0x2" ::: "memory")
#define S_WAIT_ASYNC_0()   asm volatile("s_wait_asynccnt 0x0" ::: "memory")

// ---- WMMA fragment helpers ------------------------------------------------
// A (16x32 f16): lane = row (lane&15); elems 0..7 at K=g, 8..15 at K=g+16,
//                g = (lane<16 ? 0 : 8)
// B (32x16 f16) from N-major source: lane = col; elems 0..7 at K=g,
//                8..15 at K=g+16 (same g as A)
// C (16x16 f32): value r at M = r + (lane<16?0:8), N = lane&15

static __device__ inline v16h frag_from_f32(const float* p0, const float* p1) {
  float4 a0 = *(const float4*)(p0);
  float4 a1 = *(const float4*)(p0 + 4);
  float4 b0 = *(const float4*)(p1);
  float4 b1 = *(const float4*)(p1 + 4);
  v16h r;
  r[0]=(_Float16)a0.x;  r[1]=(_Float16)a0.y;  r[2]=(_Float16)a0.z;  r[3]=(_Float16)a0.w;
  r[4]=(_Float16)a1.x;  r[5]=(_Float16)a1.y;  r[6]=(_Float16)a1.z;  r[7]=(_Float16)a1.w;
  r[8]=(_Float16)b0.x;  r[9]=(_Float16)b0.y;  r[10]=(_Float16)b0.z; r[11]=(_Float16)b0.w;
  r[12]=(_Float16)b1.x; r[13]=(_Float16)b1.y; r[14]=(_Float16)b1.z; r[15]=(_Float16)b1.w;
  return r;
}

static __device__ inline v16h frag_from_f16(const _Float16* p0, const _Float16* p1) {
  v8h lo = *(const v8h*)p0;
  v8h hi = *(const v8h*)p1;
  v16h r;
#pragma unroll
  for (int e = 0; e < 8; ++e) { r[e] = lo[e]; r[8 + e] = hi[e]; }
  return r;
}

static __device__ inline v8f wmma_f16(v16h a, v16h b, v8f c) {
  return __builtin_amdgcn_wmma_f32_16x16x32_f16(false, a, false, b,
                                                (short)0, c, false, false);
}

// ---------------------------------------------------------------------------
// GEMM: Y = A @ W^T + bias.  A is [4096,1024] (f32 or f16), W f32 [1024,1024].
// Block: 256 threads / 8 waves. Block tile: 128 rows x 64 cols.
//   wave w -> mtile = blockIdx.y*8 + w, all 4 ntiles of the block's 64 cols.
// Weight panel (64 rows x 32 k, f32, 8KB) async-staged to LDS, double buffered.
// MODE 0: f16 out, head layout [B,H,L,dk]   (Qh / Kh), A = f32
// MODE 1: f16 out, transposed  [B,H,dk,L]   (Vt),      A = f32
// MODE 2: f32 out + bias, flat [4096,1024],            A = f16
// ---------------------------------------------------------------------------
template <int MODE>
__global__ void mha_gemm(const float* __restrict__ Af32,
                         const _Float16* __restrict__ Af16,
                         const float* __restrict__ W,
                         const float* __restrict__ bias,
                         _Float16* __restrict__ outh,
                         float* __restrict__ outf) {
  extern __shared__ float Bs[];                 // [2][64*32] f32 = 16 KB
  const int tid  = threadIdx.x;
  const int lane = tid & 31;
  const int wave = tid >> 5;
  const int n0    = blockIdx.x * 64;            // 16 n-blocks
  const int mtile = blockIdx.y * 8 + wave;      // 32 m-blocks * 8 waves
  const int aG = (lane < 16) ? 0 : 8;
  const int bK = (lane < 16) ? 0 : 16;
  const int m  = mtile * 16 + (lane & 15);

  const float*    arow32 = (MODE != 2) ? Af32 + (size_t)m * D_ : nullptr;
  const _Float16* arow16 = (MODE == 2) ? Af16 + (size_t)m * D_ : nullptr;

  const unsigned ldsBase = (unsigned)(size_t)(void*)Bs;   // LDS byte offset

  // stage one 64x32 f32 weight panel into LDS buffer `buf`
  auto stage = [&](int buf, int k0) {
#pragma unroll
    for (int it = 0; it < 2; ++it) {
      const int c   = tid + it * 256;           // 512 x b128 chunks
      const int row = c >> 3, part = c & 7;
      const float* g = W + (size_t)(n0 + row) * D_ + k0 + part * 4;
      const unsigned l = ldsBase + (unsigned)(buf * 2048 + row * 32 + part * 4) * 4u;
      async_load_b128(l, g);
    }
  };

  stage(0, 0);
  stage(1, 32);

  v8f c0 = {}, c1 = {}, c2 = {}, c3 = {};
  for (int k0 = 0; k0 < D_; k0 += 32) {
    const int buf = (k0 >> 5) & 1;
    if (k0 + 32 >= D_) { S_WAIT_ASYNC_0(); } else { S_WAIT_ASYNC_LE2(); }
    __syncthreads();                            // panel `buf` visible to all

    v16h a;
    if (MODE == 2)
      a = frag_from_f16(arow16 + k0 + aG, arow16 + k0 + aG + 16);
    else
      a = frag_from_f32(arow32 + k0 + aG, arow32 + k0 + aG + 16);

    const float* bp = Bs + buf * 2048;          // [64][32] f32, N-major
#pragma unroll
    for (int t = 0; t < 4; ++t) {
      const float* row = bp + (t * 16 + (lane & 15)) * 32 + aG;
      v16h b = frag_from_f32(row, row + 16);
      if (t == 0) c0 = wmma_f16(a, b, c0);
      if (t == 1) c1 = wmma_f16(a, b, c1);
      if (t == 2) c2 = wmma_f16(a, b, c2);
      if (t == 3) c3 = wmma_f16(a, b, c3);
    }
    __syncthreads();                            // all readers done with `buf`
    if (k0 + 64 < D_) stage(buf, k0 + 64);
  }

  const int rbase = (lane < 16) ? 0 : 8;
#pragma unroll
  for (int t = 0; t < 4; ++t) {
    const v8f& c = (t == 0) ? c0 : (t == 1) ? c1 : (t == 2) ? c2 : c3;
    const int n  = n0 + t * 16 + (lane & 15);
    const float bv = bias[n];
    const int h = n >> 6, d = n & 63;
#pragma unroll
    for (int r = 0; r < 8; ++r) {
      const int i  = mtile * 16 + rbase + r;    // row in [0,4096)
      const int bb = i >> 10, l = i & 1023;
      const float val = c[r] + bv;
      if (MODE == 0)
        outh[((size_t)(bb * 16 + h) * 1024 + l) * 64 + d] = (_Float16)val;
      else if (MODE == 1)
        outh[((size_t)(bb * 16 + h) * 64 + d) * 1024 + l] = (_Float16)val;
      else
        outf[(size_t)i * D_ + n] = val;
    }
  }
}

// ---------------------------------------------------------------------------
// Attention: per (b,h, 16-query tile): S = QK^T/8 -> mask -> softmax ->
// attn (fp32, 256MB) -> P@V -> X2 f16 [B,L,D].
// grid (64,64), block 256. LDS: S[16][1032] f32 + P[16][1040] f16.
// Phase 3 uses all 8 waves: 4 d-tiles x 2 causal-range halves, LDS reduce.
// ---------------------------------------------------------------------------
__global__ void mha_attention(const _Float16* __restrict__ Qh,
                              const _Float16* __restrict__ Kh,
                              const _Float16* __restrict__ Vt,
                              const unsigned char* __restrict__ pad,
                              float* __restrict__ attn_out,
                              _Float16* __restrict__ X2) {
  extern __shared__ char smem_raw[];
  constexpr int SROW = 1032;
  constexpr int PROW = 1040;
  float*    S = (float*)smem_raw;                          // 16*SROW f32
  _Float16* P = (_Float16*)(smem_raw + 16 * SROW * 4);     // 16*PROW f16

  const int bh    = blockIdx.x;
  const int b     = bh >> 4;
  const int h     = bh & 15;
  const int mtile = blockIdx.y;
  const int lane  = threadIdx.x & 31;
  const int wave  = threadIdx.x >> 5;
  const int aG = (lane < 16) ? 0 : 8;
  const int bK = (lane < 16) ? 0 : 16;
  const int mrow = lane & 15;

  const _Float16* Qbase = Qh + ((size_t)bh * 1024 + mtile * 16) * 64;
  const _Float16* Kbase = Kh + (size_t)bh * 1024 * 64;
  const unsigned char* padrow = pad + b * 1024;

  // hoisted Q fragments (kk = 0 and 32), live across the whole jt loop
  const v16h qa0 = frag_from_f16(Qbase + mrow * 64 + aG,
                                 Qbase + mrow * 64 + aG + 16);
  const v16h qa1 = frag_from_f16(Qbase + mrow * 64 + 32 + aG,
                                 Qbase + mrow * 64 + 32 + aG + 16);

  // ---- phase 1: scores for causally-reachable key tiles --------------------
  for (int jt = wave; jt <= mtile; jt += 8) {
    const _Float16* krow = Kbase + (size_t)(jt * 16 + mrow) * 64;
    v8f c = {};
    c = wmma_f16(qa0, frag_from_f16(krow + aG, krow + aG + 16), c);
    c = wmma_f16(qa1, frag_from_f16(krow + 32 + aG, krow + 32 + aG + 16), c);

    const int j = jt * 16 + (lane & 15);
    const bool padj = padrow[j] != 0;
    const int rbase = (lane < 16) ? 0 : 8;
#pragma unroll
    for (int r = 0; r < 8; ++r) {
      const int i = mtile * 16 + rbase + r;
      float v = c[r] * 0.125f;                 // 1/sqrt(d_k)
      if (j > i || padj) v = NEG_INF_F;
      S[(rbase + r) * SROW + j] = v;
    }
  }
  __syncthreads();

  // ---- phase 2: softmax (16 threads per row); cache exp() in S -------------
  {
    const int r   = threadIdx.x >> 4;
    const int l16 = threadIdx.x & 15;
    const int i   = mtile * 16 + r;

    float mx = NEG_INF_F;
    for (int cc = l16; cc <= i; cc += 16) mx = fmaxf(mx, S[r * SROW + cc]);
#pragma unroll
    for (int off = 8; off >= 1; off >>= 1) mx = fmaxf(mx, __shfl_xor(mx, off, 16));

    float sum = 0.f;
    for (int cc = l16; cc <= i; cc += 16) {
      const float e = __expf(S[r * SROW + cc] - mx);
      S[r * SROW + cc] = e;
      sum += e;
    }
#pragma unroll
    for (int off = 8; off >= 1; off >>= 1) sum += __shfl_xor(sum, off, 16);
    const float inv = 1.0f / sum;

    float* attn_row = attn_out + ((size_t)bh * 1024 + i) * 1024;
    for (int cc = l16; cc < 1024; cc += 16) {
      const float pr = (cc <= i) ? S[r * SROW + cc] * inv : 0.0f;
      attn_row[cc]     = pr;
      P[r * PROW + cc] = (_Float16)pr;
    }
  }
  __syncthreads();

  // ---- phase 3: out_h = P @ V; 8 waves = 4 d-tiles x 2 k-range halves ------
  const int dtile = wave & 3;
  const int half  = wave >> 2;
  const int jsteps = mtile / 2 + 1;            // 32-wide K steps to frontier
  const int hs = (jsteps + 1) >> 1;
  const int s0 = half ? hs : 0;
  const int s1 = half ? jsteps : hs;

  const _Float16* Vbase =
      Vt + ((size_t)bh * 64 + dtile * 16 + (lane & 15)) * 1024;
  v8f c = {};
  for (int s = s0; s < s1; ++s) {
    const int kk = s * 32;
    v16h a  = frag_from_f16(P + mrow * PROW + kk + aG,
                            P + mrow * PROW + kk + aG + 16);
    v16h bb = frag_from_f16(Vbase + kk + aG, Vbase + kk + aG + 16);
    c = wmma_f16(a, bb, c);
  }

  const int rbase = (lane < 16) ? 0 : 8;
  float* R = S;                                 // reuse S for partial sums
  if (half == 1) {
#pragma unroll
    for (int r = 0; r < 8; ++r)
      R[(dtile * 16 + rbase + r) * 16 + (lane & 15)] = c[r];
  }
  __syncthreads();
  if (half == 0) {
    const int ncol = h * 64 + dtile * 16 + (lane & 15);
#pragma unroll
    for (int r = 0; r < 8; ++r) {
      const int i = mtile * 16 + rbase + r;
      const float v = c[r] + R[(dtile * 16 + rbase + r) * 16 + (lane & 15)];
      X2[((size_t)b * 1024 + i) * 1024 + ncol] = (_Float16)v;
    }
  }
}

// ---------------------------------------------------------------------------
extern "C" void kernel_launch(void* const* d_in, const int* in_sizes, int n_in,
                              void* d_out, int out_size, void* d_ws, size_t ws_size,
                              hipStream_t stream) {
  (void)in_sizes; (void)n_in; (void)out_size; (void)ws_size;

  const float* q  = (const float*)d_in[0];
  const float* k  = (const float*)d_in[1];
  const float* v  = (const float*)d_in[2];
  const unsigned char* pad = (const unsigned char*)d_in[3];
  const float* Wq = (const float*)d_in[4];
  const float* bq = (const float*)d_in[5];
  const float* Wk = (const float*)d_in[6];
  const float* bk = (const float*)d_in[7];
  const float* Wv = (const float*)d_in[8];
  const float* bv = (const float*)d_in[9];
  const float* Wo = (const float*)d_in[10];
  const float* bo = (const float*)d_in[11];

  float* out  = (float*)d_out;                         // [4,1024,1024]
  float* attn = out + (size_t)B_ * L_ * D_;            // [4,16,1024,1024]

  const size_t HEAD_ELEMS = (size_t)B_ * H_ * L_ * DK_;   // 4,194,304
  _Float16* Qh = (_Float16*)d_ws;
  _Float16* Kh = Qh + HEAD_ELEMS;
  _Float16* Vt = Kh + HEAD_ELEMS;
  _Float16* X2 = Vt + HEAD_ELEMS;                      // total 32 MB f16

  dim3 blk(256);
  dim3 ggemm(D_ / 64, (B_ * L_) / 128);                // (16, 32)
  const size_t smem_gemm = 2 * 64 * 32 * sizeof(float);  // 16 KB

  mha_gemm<0><<<ggemm, blk, smem_gemm, stream>>>(q, nullptr, Wq, bq, Qh, nullptr);
  mha_gemm<0><<<ggemm, blk, smem_gemm, stream>>>(k, nullptr, Wk, bk, Kh, nullptr);
  mha_gemm<1><<<ggemm, blk, smem_gemm, stream>>>(v, nullptr, Wv, bv, Vt, nullptr);

  const size_t smem_attn = 16 * 1032 * sizeof(float) + 16 * 1040 * sizeof(_Float16);
  mha_attention<<<dim3(B_ * H_, L_ / 16), blk, smem_attn, stream>>>(
      Qh, Kh, Vt, pad, attn, X2);

  mha_gemm<2><<<ggemm, blk, smem_gemm, stream>>>(nullptr, X2, Wo, bo, nullptr, out);
}
